// MultiHeadSelfAttention_33105607917939
// MI455X (gfx1250) — matrix operand order repaired and checked by reading, not compiled, not run on hardware
//
#include <hip/hip_runtime.h>

#define EMBED 192
#define NHEAD 4
#define HDIM  48
#define DPAD  64
#define BB    8
#define SS    2048
#define BS    (BB*SS)   // 16384 rows total

typedef __attribute__((ext_vector_type(16))) __bf16 v16bf;
typedef __attribute__((ext_vector_type(8)))  float  v8f;

union Frag { unsigned int u[8]; v16bf v; };

__device__ __forceinline__ unsigned short f2bf(float f) {
    union { __bf16 b; unsigned short s; } r; r.b = (__bf16)f; return r.s;
}
__device__ __forceinline__ unsigned int pk_bf(float lo, float hi) {
    union { __bf16 b[2]; unsigned int u; } r;
    r.b[0] = (__bf16)lo; r.b[1] = (__bf16)hi; return r.u;
}
// A-fragment K index for 16-bit A (16x32): lanes 0-15 hold K {0..7,16..23}, lanes 16-31 {8..15,24..31}
__device__ __forceinline__ int kA(int v, int half) { return 2*(v&3) + 16*(v>>2) + 8*half; }
// B-fragment K index (32x16): lanes 0-15 hold K {0..7,16..23}, lanes 16-31 {8..15,24..31}
__device__ __forceinline__ int kB(int j, int half) { return 2*(j&3) + 16*(j>>2) + 8*half; }

__device__ __forceinline__ v8f wmma_bf16(const Frag& a, const Frag& b, v8f c) {
    return __builtin_amdgcn_wmma_f32_16x16x32_bf16(false, a.v, false, b.v,
                                                   (short)0, c, false, false);
}

// ---------------------------------------------------------------------------
// Prep: transpose + convert weights to bf16, [out][in] layout (K-contiguous B)
// ---------------------------------------------------------------------------
__global__ void prep_weights(const float* __restrict__ Wq, const float* __restrict__ Wk,
                             const float* __restrict__ Wv, const float* __restrict__ Wo,
                             unsigned short* __restrict__ WT4) {
    int idx = blockIdx.x * blockDim.x + threadIdx.x;
    if (idx >= 4*EMBED*EMBED) return;
    int m = idx / (EMBED*EMBED);
    int r = idx % (EMBED*EMBED);
    int o = r / EMBED, i = r % EMBED;
    const float* W = (m==0)?Wq:(m==1)?Wk:(m==2)?Wv:Wo;
    WT4[idx] = f2bf(W[i*EMBED + o]);   // WT[m][o][i] = W[i][o]
}

// ---------------------------------------------------------------------------
// QKV projection: one wave computes a 16x192 strip of Q, K or V.
//   Q,K -> [B,H,S,DPAD] bf16 (pad zeroed), Q pre-scaled by 1/sqrt(48)
//   V   -> [B,H,HDIM,S] bf16 (transposed so P·V B-operand is K-contiguous)
// ---------------------------------------------------------------------------
__global__ void __launch_bounds__(32)
qkv_kernel(const float* __restrict__ hidden, const unsigned short* __restrict__ WT4,
           const float* __restrict__ bq, const float* __restrict__ bk,
           const float* __restrict__ bv,
           unsigned short* __restrict__ Qb, unsigned short* __restrict__ Kb,
           unsigned short* __restrict__ VT)
{
    const int lane = threadIdx.x & 31;
    const int lo = lane & 15, half = lane >> 4;
    const int tile = blockIdx.x;          // 0..1023
    const int mat  = blockIdx.y;          // 0=Q,1=K,2=V
    const int bs0  = tile * 16;
    const int b    = bs0 / SS;
    const int s0   = bs0 % SS;

    const unsigned short* WT = WT4 + (size_t)mat * (EMBED*EMBED);
    const float* bias = (mat==0)?bq:(mat==1)?bk:bv;
    const float qscale = (mat==0) ? 0.14433756729740643f /* 1/sqrt(48) */ : 1.0f;

    // A fragments: 16 x 192 of hidden, f32 -> bf16 packed on the fly
    Frag av[6];
    const float* arow = hidden + (size_t)(bs0 + lo) * EMBED;
    #pragma unroll
    for (int ks = 0; ks < 6; ++ks)
        #pragma unroll
        for (int v = 0; v < 8; ++v) {
            int kk = ks*32 + kA(v, half);
            float2 f = *(const float2*)(arow + kk);
            av[ks].u[v] = pk_bf(f.x, f.y);
        }

    #pragma unroll 1
    for (int nt = 0; nt < 12; ++nt) {
        float bval = bias[nt*16 + lo];
        v8f c;
        #pragma unroll
        for (int r = 0; r < 8; ++r) c[r] = bval;
        const unsigned short* wrow = WT + (size_t)(nt*16 + lo) * EMBED;
        #pragma unroll
        for (int ks = 0; ks < 6; ++ks) {
            Frag bfr;
            #pragma unroll
            for (int j = 0; j < 8; ++j)
                bfr.u[j] = *(const unsigned int*)(wrow + ks*32 + kB(j, half));
            c = wmma_bf16(av[ks], bfr, c);
        }
        const int h  = nt / 3;
        const int d0 = (nt % 3) * 16;
        const int bh = b * NHEAD + h;
        if (mat < 2) {
            unsigned short* dst = (mat==0) ? Qb : Kb;
            #pragma unroll
            for (int r = 0; r < 8; ++r) {
                int srow = s0 + r + 8*half;
                dst[((size_t)bh*SS + srow)*DPAD + d0 + lo] = f2bf(c[r] * qscale);
            }
        } else {
            #pragma unroll
            for (int r = 0; r < 8; ++r) {
                int srow = s0 + r + 8*half;
                VT[((size_t)bh*HDIM + d0 + lo)*SS + srow] = f2bf(c[r]);
            }
        }
    }

    // zero pad region d = 48..63 of Q / K (so K=64 score WMMAs are exact)
    if (mat < 2) {
        unsigned short* dst = (mat==0) ? Qb : Kb;
        #pragma unroll 1
        for (int i = 0; i < 16; ++i) {
            int idx = i*32 + lane;          // 0..511 = 4 heads * 16 rows * 8 chunks
            int hh = idx & 3;
            int rr = (idx >> 2) & 15;
            int ch = (idx >> 6) & 7;
            unsigned int* p = (unsigned int*)
                &dst[(((size_t)(b*NHEAD + hh))*SS + s0 + rr)*DPAD + HDIM + ch*2];
            *p = 0u;
        }
    }
}

// ---------------------------------------------------------------------------
// Flash attention: one wave handles 16 queries of one (b,h); 64-key chunks,
// online softmax; P re-laid-out C-frag -> A-frag through padded LDS.
// ---------------------------------------------------------------------------
__global__ void __launch_bounds__(32)
attn_kernel(const unsigned short* __restrict__ Qb,
            const unsigned short* __restrict__ Kb,
            const unsigned short* __restrict__ VT,
            const float* __restrict__ maskp,
            unsigned short* __restrict__ attnB)
{
    __shared__ unsigned short Pld[16*68];   // stride 68 avoids LDS bank conflicts
    const int lane = threadIdx.x & 31;
    const int lo = lane & 15, half = lane >> 4;
    const int bh = blockIdx.y;              // 0..31
    const int b = bh >> 2, h = bh & 3;
    const int q0 = blockIdx.x * 16;

    // Q fragments: 16 x 64 (already scaled by 1/sqrt(D))
    Frag aq[2];
    const unsigned short* qrow = Qb + ((size_t)bh*SS + q0 + lo)*DPAD;
    #pragma unroll
    for (int ks = 0; ks < 2; ++ks)
        #pragma unroll
        for (int v = 0; v < 8; ++v)
            aq[ks].u[v] = *(const unsigned int*)(qrow + ks*32 + kA(v, half));

    float m_i[8], l_i[8];
    v8f acc[3];
    #pragma unroll
    for (int r = 0; r < 8; ++r) { m_i[r] = -__builtin_inff(); l_i[r] = 0.f; }
    #pragma unroll
    for (int d = 0; d < 3; ++d)
        #pragma unroll
        for (int r = 0; r < 8; ++r) acc[d][r] = 0.f;

    const float* mrow = maskp + (size_t)b * SS;

    for (int kc = 0; kc < SS/64; ++kc) {
        const int k0 = kc*64;
        // ---- scores S = Q·K^T + mask : 4 n-tiles x 2 k-steps -------------
        float sc[4][8];
        #pragma unroll
        for (int nt = 0; nt < 4; ++nt) {
            v8f c;
            #pragma unroll
            for (int r = 0; r < 8; ++r) c[r] = 0.f;
            const unsigned short* krow = Kb + ((size_t)bh*SS + k0 + nt*16 + lo)*DPAD;
            #pragma unroll
            for (int ks = 0; ks < 2; ++ks) {
                Frag bk;
                #pragma unroll
                for (int j = 0; j < 8; ++j)
                    bk.u[j] = *(const unsigned int*)(krow + ks*32 + kB(j, half));
                c = wmma_bf16(aq[ks], bk, c);
            }
            float mk = mrow[k0 + nt*16 + lo];   // additive mask per key (lane = key)
            #pragma unroll
            for (int r = 0; r < 8; ++r) sc[nt][r] = c[r] + mk;
        }
        // ---- online softmax update (row stats via cross-lane reductions) -
        float mnew[8], alpha[8];
        #pragma unroll
        for (int r = 0; r < 8; ++r) {
            float mx = fmaxf(fmaxf(sc[0][r], sc[1][r]), fmaxf(sc[2][r], sc[3][r]));
            #pragma unroll
            for (int off = 1; off < 16; off <<= 1)
                mx = fmaxf(mx, __shfl_xor(mx, off, 32));
            mnew[r] = fmaxf(mx, m_i[r]);
        }
        #pragma unroll
        for (int r = 0; r < 8; ++r) {
            alpha[r] = __expf(m_i[r] - mnew[r]);
            m_i[r] = mnew[r];
            float s = 0.f;
            #pragma unroll
            for (int nt = 0; nt < 4; ++nt) {
                float pv = __expf(sc[nt][r] - mnew[r]);
                sc[nt][r] = pv;
                s += pv;
            }
            #pragma unroll
            for (int off = 1; off < 16; off <<= 1)
                s += __shfl_xor(s, off, 32);
            l_i[r] = l_i[r]*alpha[r] + s;
        }
        // ---- P (C-frag layout) -> LDS -> A-frag layout -------------------
        #pragma unroll
        for (int nt = 0; nt < 4; ++nt)
            #pragma unroll
            for (int r = 0; r < 8; ++r)
                Pld[(r + 8*half)*68 + nt*16 + lo] = f2bf(sc[nt][r]);
        __syncthreads();
        Frag ap[2];
        #pragma unroll
        for (int ks = 0; ks < 2; ++ks)
            #pragma unroll
            for (int v = 0; v < 8; ++v)
                ap[ks].u[v] = *(const unsigned int*)&Pld[lo*68 + ks*32 + kA(v, half)];
        // ---- O = alpha*O + P·V : 3 n-tiles (d) x 2 k-steps ---------------
        #pragma unroll
        for (int d = 0; d < 3; ++d) {
            #pragma unroll
            for (int r = 0; r < 8; ++r) acc[d][r] *= alpha[r];
            const unsigned short* vrow = VT + ((size_t)bh*HDIM + d*16 + lo)*SS + k0;
            #pragma unroll
            for (int ks = 0; ks < 2; ++ks) {
                Frag bv;
                #pragma unroll
                for (int j = 0; j < 8; ++j)
                    bv.u[j] = *(const unsigned int*)(vrow + ks*32 + kB(j, half));
                acc[d] = wmma_bf16(ap[ks], bv, acc[d]);
            }
        }
        __syncthreads();
    }
    // ---- normalize and store merged-head output (bf16) -------------------
    #pragma unroll
    for (int d = 0; d < 3; ++d)
        #pragma unroll
        for (int r = 0; r < 8; ++r) {
            int srow = q0 + r + 8*half;
            float o = acc[d][r] / l_i[r];
            attnB[((size_t)b*SS + srow)*EMBED + h*HDIM + d*16 + lo] = f2bf(o);
        }
}

// ---------------------------------------------------------------------------
// Output projection: attnB(bf16) x WoT + bo -> fp32 out
// ---------------------------------------------------------------------------
__global__ void __launch_bounds__(32)
outproj_kernel(const unsigned short* __restrict__ attnB,
               const unsigned short* __restrict__ WoT,
               const float* __restrict__ bo,
               float* __restrict__ out)
{
    const int lane = threadIdx.x & 31;
    const int lo = lane & 15, half = lane >> 4;
    const int bs0 = blockIdx.x * 16;

    Frag av[6];
    const unsigned short* arow = attnB + (size_t)(bs0 + lo) * EMBED;
    #pragma unroll
    for (int ks = 0; ks < 6; ++ks)
        #pragma unroll
        for (int v = 0; v < 8; ++v)
            av[ks].u[v] = *(const unsigned int*)(arow + ks*32 + kA(v, half));

    #pragma unroll 1
    for (int nt = 0; nt < 12; ++nt) {
        float bval = bo[nt*16 + lo];
        v8f c;
        #pragma unroll
        for (int r = 0; r < 8; ++r) c[r] = bval;
        const unsigned short* wrow = WoT + (size_t)(nt*16 + lo) * EMBED;
        #pragma unroll
        for (int ks = 0; ks < 6; ++ks) {
            Frag bfr;
            #pragma unroll
            for (int j = 0; j < 8; ++j)
                bfr.u[j] = *(const unsigned int*)(wrow + ks*32 + kB(j, half));
            c = wmma_bf16(av[ks], bfr, c);
        }
        #pragma unroll
        for (int r = 0; r < 8; ++r)
            out[(size_t)(bs0 + r + 8*half)*EMBED + nt*16 + lo] = c[r];
    }
}

// ---------------------------------------------------------------------------
extern "C" void kernel_launch(void* const* d_in, const int* in_sizes, int n_in,
                              void* d_out, int out_size, void* d_ws, size_t ws_size,
                              hipStream_t stream) {
    (void)in_sizes; (void)n_in; (void)out_size; (void)ws_size;
    const float* hidden = (const float*)d_in[0];
    const float* mask   = (const float*)d_in[1];
    const float* Wq = (const float*)d_in[2];
    const float* bq = (const float*)d_in[3];
    const float* Wk = (const float*)d_in[4];
    const float* bk = (const float*)d_in[5];
    const float* Wv = (const float*)d_in[6];
    const float* bv = (const float*)d_in[7];
    const float* Wo = (const float*)d_in[8];
    const float* bo = (const float*)d_in[9];
    float* out = (float*)d_out;

    char* p = (char*)d_ws;
    unsigned short* WT4   = (unsigned short*)p; p += (size_t)4*EMBED*EMBED*2;     // 288 KB
    unsigned short* Qb    = (unsigned short*)p; p += (size_t)BB*NHEAD*SS*DPAD*2;  // 8 MB
    unsigned short* Kb    = (unsigned short*)p; p += (size_t)BB*NHEAD*SS*DPAD*2;  // 8 MB
    unsigned short* VT    = (unsigned short*)p; p += (size_t)BB*NHEAD*HDIM*SS*2;  // 6 MB
    unsigned short* attnB = (unsigned short*)p; p += (size_t)BS*EMBED*2;          // 6 MB

    prep_weights<<<(4*EMBED*EMBED + 255)/256, 256, 0, stream>>>(Wq, Wk, Wv, Wo, WT4);
    qkv_kernel<<<dim3(BS/16, 3), 32, 0, stream>>>(hidden, WT4, bq, bk, bv, Qb, Kb, VT);
    attn_kernel<<<dim3(SS/16, BB*NHEAD), 32, 0, stream>>>(Qb, Kb, VT, mask, attnB);
    outproj_kernel<<<BS/16, 32, 0, stream>>>(attnB, WT4 + 3*EMBED*EMBED, bo, out);
}
